// MolGenTransformerWrapper_60945585930356
// MI455X (gfx1250) — matrix operand order repaired and checked by reading, not compiled, not run on hardware
//
#include <hip/hip_runtime.h>
#include <hip/hip_bf16.h>

typedef __attribute__((ext_vector_type(16))) _Float16 v16h;
typedef __attribute__((ext_vector_type(8)))  float    v8f;

#define BB     8
#define LL     24
#define DD     256
#define HH     8
#define HDD    32
#define FFF    1024
#define VV     64
#define NLL    2
#define NSTEPS 23
#define NROWS  (BB * LL)   // 192

// ---------------------------------------------------------------------------
// Core WMMA tile: acc += Ah[16 x K] (f16 row-major, lda) * Bt[16 x K]^T
// (f16 weights stored transposed: Bt[n][k]).  K multiple of 32.
// A layout (16-bit 16x32): lanes 0-15 M=lane, elems 0..7 = K 0..7, 8..15 = K 16..23;
//                          lanes 16-31 same M, K 8..15 / 24..31.
// B layout (32x16): same K order as A: lanes 0-15 K 0..7 / 16..23, lanes 16-31 K 8..15 / 24..31.
// Per lane all four 8-half runs are K-contiguous -> b128 loads.
// ---------------------------------------------------------------------------
__device__ inline v8f wmma_gemm_acc(const _Float16* __restrict__ Ah, int lda,
                                    const _Float16* __restrict__ Bt, int ldb,
                                    int K, v8f acc) {
  const int lane  = threadIdx.x & 31;
  const int half  = lane >> 4;
  const int mr    = lane & 15;
  const int aoff  = half << 3;     // K offset of first A run
  const int boff  = half << 3;     // K offset of first B run
  for (int kb = 0; kb < K; kb += 32) {
    v16h a, b;
#pragma unroll
    for (int e = 0; e < 8; ++e) {
      a[e]     = Ah[mr * lda + kb + aoff + e];
      a[e + 8] = Ah[mr * lda + kb + 16 + aoff + e];
      b[e]     = Bt[mr * ldb + kb + boff + e];
      b[e + 8] = Bt[mr * ldb + kb + 16 + boff + e];
    }
    acc = __builtin_amdgcn_wmma_f32_16x16x32_f16(false, a, false, b,
                                                 (short)0, acc, false, false);
  }
  return acc;
}

// ---------------------------------------------------------------------------
// Setup kernels
// ---------------------------------------------------------------------------
__global__ void pe_kernel(float* __restrict__ pe) {
  int l = blockIdx.x;           // 0..23
  int i = threadIdx.x;          // 0..127
  float div = __expf((2.0f * i) * (-logf(10000.0f) / (float)DD));
  float a = (float)l * div;
  pe[l * DD + 2 * i]     = __sinf(a);
  pe[l * DD + 2 * i + 1] = __cosf(a);
}

__global__ void init_tok_kernel(int* __restrict__ tok) {
  int i = threadIdx.x;
  if (i < NROWS) tok[i] = ((i % LL) == 0) ? 1 /*SOS*/ : 0 /*PAD*/;
}

// fp32 [K,N] row-major  ->  f16 transposed [N,K]
__global__ void convt_kernel(const float* __restrict__ W, _Float16* __restrict__ Wt,
                             int K, int N) {
  int idx = blockIdx.x * blockDim.x + threadIdx.x;
  if (idx >= K * N) return;
  int k = idx / N, n = idx % N;
  Wt[n * K + k] = (_Float16)W[idx];
}

// fp32 -> f16 straight copy (activation sidecar)
__global__ void cvt_h_kernel(const float* __restrict__ src, _Float16* __restrict__ dst,
                             int n) {
  int idx = blockIdx.x * blockDim.x + threadIdx.x;
  if (idx < n) dst[idx] = (_Float16)src[idx];
}

// ---------------------------------------------------------------------------
// Generic GEMM: C[M,N] = Ah[M,K] @ W + bias (optional ReLU).  Wave per 16x16
// tile.  Writes f16 (Ch != null) or f32 (Cf) output.
// ---------------------------------------------------------------------------
__global__ __launch_bounds__(256)
void gemm_kernel(const _Float16* __restrict__ Ah, const _Float16* __restrict__ Bt,
                 const float* __restrict__ bias, float* __restrict__ Cf,
                 _Float16* __restrict__ Ch, int M, int N, int K, int relu) {
  int wave    = threadIdx.x >> 5;
  int tile    = blockIdx.x * 8 + wave;
  int ntilesN = N >> 4;
  int ntiles  = (M >> 4) * ntilesN;
  if (tile >= ntiles) return;                 // wave-uniform: EXEC stays full
  int m0 = (tile / ntilesN) << 4;
  int n0 = (tile % ntilesN) << 4;
  v8f acc = {};
  acc = wmma_gemm_acc(Ah + m0 * K, K, Bt + n0 * K, K, K, acc);
  int lane = threadIdx.x & 31;
  int col = lane & 15, rbase = (lane >> 4) << 3;
  float bv = bias[n0 + col];
#pragma unroll
  for (int r = 0; r < 8; ++r) {
    float v = acc[r] + bv;
    if (relu) v = fmaxf(v, 0.0f);
    if (Ch) Ch[(m0 + rbase + r) * N + n0 + col] = (_Float16)v;
    else    Cf[(m0 + rbase + r) * N + n0 + col] = v;
  }
}

// ---------------------------------------------------------------------------
// Fused Q/K/V projections: one launch, tile space = 3 x (M/16 x N/16).
// ---------------------------------------------------------------------------
__global__ __launch_bounds__(256)
void qkv_kernel(const _Float16* __restrict__ Ah,
                const _Float16* __restrict__ Btq, const _Float16* __restrict__ Btk,
                const _Float16* __restrict__ Btv,
                const float* __restrict__ bq, const float* __restrict__ bk,
                const float* __restrict__ bv,
                float* __restrict__ q, float* __restrict__ k, float* __restrict__ v) {
  const int K = DD, N = DD;
  int wave = threadIdx.x >> 5;
  int tile = blockIdx.x * 8 + wave;
  const int perMat = (NROWS / 16) * (N / 16);   // 192
  if (tile >= 3 * perMat) return;
  int mat = tile / perMat;
  int sub = tile % perMat;
  int m0 = (sub / (N / 16)) << 4;
  int n0 = (sub % (N / 16)) << 4;
  const _Float16* Bt; const float* bias; float* C;
  if (mat == 0)      { Bt = Btq; bias = bq; C = q; }
  else if (mat == 1) { Bt = Btk; bias = bk; C = k; }
  else               { Bt = Btv; bias = bv; C = v; }
  v8f acc = {};
  acc = wmma_gemm_acc(Ah + m0 * K, K, Bt + n0 * K, K, K, acc);
  int lane = threadIdx.x & 31;
  int col = lane & 15, rbase = (lane >> 4) << 3;
  float bvv = bias[n0 + col];
#pragma unroll
  for (int r = 0; r < 8; ++r)
    C[(m0 + rbase + r) * N + n0 + col] = acc[r] + bvv;
}

// ---------------------------------------------------------------------------
// Fused: out = LayerNorm(resid + Ah@W + bias).  N == DD == 256.
// Block = 512 threads (16 waves); wave w computes N-tile w, then LN of row w.
// Writes f32 out and f16 sidecar outh.
// ---------------------------------------------------------------------------
__global__ __launch_bounds__(512)
void gemm_res_ln_kernel(const _Float16* __restrict__ Ah, const _Float16* __restrict__ Bt,
                        const float* __restrict__ bias, const float* __restrict__ resid,
                        const float* __restrict__ g, const float* __restrict__ bb,
                        float* __restrict__ out, _Float16* __restrict__ outh, int K) {
  __shared__ float ly[16][DD];
  int wave = threadIdx.x >> 5;       // 0..15 : n-tile index, then LN row
  int lane = threadIdx.x & 31;
  int m0 = blockIdx.x << 4;
  int n0 = wave << 4;
  v8f acc = {};
  acc = wmma_gemm_acc(Ah + m0 * K, K, Bt + n0 * K, K, K, acc);
  int col = lane & 15, rbase = (lane >> 4) << 3;
  float bv = bias[n0 + col];
#pragma unroll
  for (int r = 0; r < 8; ++r) {
    int row = rbase + r;
    ly[row][n0 + col] = acc[r] + bv + resid[(m0 + row) * DD + n0 + col];
  }
  __syncthreads();
  float s = 0.0f, ss = 0.0f;
#pragma unroll
  for (int i = 0; i < 8; ++i) {
    float yv = ly[wave][lane + i * 32];
    s += yv; ss += yv * yv;
  }
#pragma unroll
  for (int o = 16; o >= 1; o >>= 1) {
    s  += __shfl_xor(s,  o, 32);
    ss += __shfl_xor(ss, o, 32);
  }
  float mean = s * (1.0f / DD);
  float var  = ss * (1.0f / DD) - mean * mean;
  float rs   = rsqrtf(var + 1e-5f);
#pragma unroll
  for (int i = 0; i < 8; ++i) {
    int d = lane + i * 32;
    float o2 = (ly[wave][d] - mean) * rs * g[d] + bb[d];
    out [(m0 + wave) * DD + d] = o2;
    outh[(m0 + wave) * DD + d] = (_Float16)o2;
  }
}

// ---------------------------------------------------------------------------
// Attention per (b,h): L=24 queries, HD=32.  K/V staged in LDS; lane = query.
// Output stored f16 (only consumed as GEMM A operand).
// ---------------------------------------------------------------------------
__global__ __launch_bounds__(32)
void attn_kernel(const float* __restrict__ q, const float* __restrict__ k,
                 const float* __restrict__ v, _Float16* __restrict__ o, int causal) {
  __shared__ float ks[LL][HDD];
  __shared__ float vs[LL][HDD];
  int b = blockIdx.x >> 3, h = blockIdx.x & 7;
  int lane = threadIdx.x;
  int cbase = h * HDD;
  for (int idx = lane; idx < LL * HDD; idx += 32) {
    int r = idx >> 5, d = idx & 31;
    ks[r][d] = k[(b * LL + r) * DD + cbase + d];
    vs[r][d] = v[(b * LL + r) * DD + cbase + d];
  }
  __syncthreads();
  if (lane < LL) {
    float qv[HDD];
#pragma unroll
    for (int d = 0; d < HDD; ++d) qv[d] = q[(b * LL + lane) * DD + cbase + d];
    const float scale = 0.17677669529663688f;   // 1/sqrt(32)
    float s[LL];
    float mx = -1e30f;
#pragma unroll
    for (int ki = 0; ki < LL; ++ki) {
      float acc = 0.0f;
#pragma unroll
      for (int d = 0; d < HDD; ++d) acc += qv[d] * ks[ki][d];
      acc *= scale;
      if (causal && ki > lane) acc += -1e9f;
      s[ki] = acc;
      mx = fmaxf(mx, acc);
    }
    float sum = 0.0f;
#pragma unroll
    for (int ki = 0; ki < LL; ++ki) { s[ki] = __expf(s[ki] - mx); sum += s[ki]; }
    float inv = 1.0f / sum;
#pragma unroll
    for (int d = 0; d < HDD; ++d) {
      float acc = 0.0f;
#pragma unroll
      for (int ki = 0; ki < LL; ++ki) acc += s[ki] * vs[ki][d];
      o[(b * LL + lane) * DD + cbase + d] = (_Float16)(acc * inv);
    }
  }
}

// ---------------------------------------------------------------------------
// x[row] = emb[tok[row]] + pe[row % L]   (f32 + f16 sidecar)
// ---------------------------------------------------------------------------
__global__ __launch_bounds__(DD)
void embed_kernel(const int* __restrict__ tok, const float* __restrict__ emb,
                  const float* __restrict__ pe, float* __restrict__ x,
                  _Float16* __restrict__ xh) {
  int row = blockIdx.x;        // b*L + l
  int d   = threadIdx.x;
  int l   = row % LL;
  int tk  = tok[row];
  float v = emb[tk * DD + d] + pe[l * DD + d];
  x [row * DD + d] = v;
  xh[row * DD + d] = (_Float16)v;
}

// ---------------------------------------------------------------------------
// logits_t = h[:,t,:] @ Wout + bout ; argmax (first max) -> next token.
// ---------------------------------------------------------------------------
__global__ __launch_bounds__(VV)
void logits_kernel(const float* __restrict__ x, const float* __restrict__ Wout,
                   const float* __restrict__ bout, int* __restrict__ tok,
                   float* __restrict__ out_logits, int* __restrict__ out_tok, int t) {
  __shared__ float lg[VV];
  int b = blockIdx.x, vt = threadIdx.x;
  const float* h = x + (b * LL + t) * DD;
  float acc = bout[vt];
  for (int d = 0; d < DD; ++d) acc += h[d] * Wout[d * VV + vt];
  lg[vt] = acc;
  out_logits[(t * BB + b) * VV + vt] = acc;
  __syncthreads();
  if (vt == 0) {
    int best = 0; float bv = lg[0];
    for (int i = 1; i < VV; ++i) if (lg[i] > bv) { bv = lg[i]; best = i; }
    tok[b * LL + t + 1] = best;
    out_tok[b * NSTEPS + t] = best;
  }
}

// ---------------------------------------------------------------------------
// Host orchestration
// ---------------------------------------------------------------------------
extern "C" void kernel_launch(void* const* d_in, const int* in_sizes, int n_in,
                              void* d_out, int out_size, void* d_ws, size_t ws_size,
                              hipStream_t stream) {
  (void)in_sizes; (void)n_in; (void)out_size; (void)ws_size;

  const float* z    = (const float*)d_in[0];
  const float* emb  = (const float*)d_in[1];
  const float* Wout = (const float*)d_in[2];
  const float* bout = (const float*)d_in[3];
  const float* sa_W[4] = { (const float*)d_in[4],  (const float*)d_in[6],
                           (const float*)d_in[8],  (const float*)d_in[10] }; // q,k,v,o
  const float* sa_b[4] = { (const float*)d_in[5],  (const float*)d_in[7],
                           (const float*)d_in[9],  (const float*)d_in[11] };
  const float* ca_W[4] = { (const float*)d_in[12], (const float*)d_in[14],
                           (const float*)d_in[16], (const float*)d_in[18] };
  const float* ca_b[4] = { (const float*)d_in[13], (const float*)d_in[15],
                           (const float*)d_in[17], (const float*)d_in[19] };
  const float* W1 = (const float*)d_in[20];
  const float* b1 = (const float*)d_in[21];
  const float* W2 = (const float*)d_in[22];
  const float* b2 = (const float*)d_in[23];
  const float* ln_g[3] = { (const float*)d_in[24], (const float*)d_in[26],
                           (const float*)d_in[28] };
  const float* ln_b[3] = { (const float*)d_in[25], (const float*)d_in[27],
                           (const float*)d_in[29] };

  // workspace bump allocator
  char* ws = (char*)d_ws;
  size_t off = 0;
  auto alloc = [&](size_t bytes) -> void* {
    void* p = ws + off;
    off = (off + bytes + 255) & ~(size_t)255;
    return p;
  };

  int*      tok  = (int*)     alloc(NROWS * sizeof(int));
  float*    pe   = (float*)   alloc(LL * DD * sizeof(float));
  float*    x    = (float*)   alloc(NROWS * DD * sizeof(float));
  _Float16* xh   = (_Float16*)alloc(NROWS * DD * sizeof(_Float16));
  float*    qb   = (float*)   alloc(NROWS * DD * sizeof(float));
  float*    kb   = (float*)   alloc(NROWS * DD * sizeof(float));
  float*    vb   = (float*)   alloc(NROWS * DD * sizeof(float));
  _Float16* abh  = (_Float16*)alloc(NROWS * DD * sizeof(_Float16));   // attn out (f16)
  _Float16* h1h  = (_Float16*)alloc(NROWS * FFF * sizeof(_Float16));  // FF hidden (f16)
  float*    kmem = (float*)   alloc(NLL * NROWS * DD * sizeof(float));
  float*    vmem = (float*)   alloc(NLL * NROWS * DD * sizeof(float));
  _Float16* zh   = (_Float16*)alloc(NROWS * DD * sizeof(_Float16));   // memory (f16)
  _Float16* sa_Wt[4], *ca_Wt[4];
  for (int m = 0; m < 4; ++m) sa_Wt[m] = (_Float16*)alloc(NLL * DD * DD * sizeof(_Float16));
  for (int m = 0; m < 4; ++m) ca_Wt[m] = (_Float16*)alloc(NLL * DD * DD * sizeof(_Float16));
  _Float16* W1t = (_Float16*)alloc(NLL * DD * FFF * sizeof(_Float16));
  _Float16* W2t = (_Float16*)alloc(NLL * FFF * DD * sizeof(_Float16));

  float* out_logits = (float*)d_out;
  int*   out_tok    = (int*)d_out + NSTEPS * BB * VV;

  // --- one-time setup (per call; deterministic) ---
  hipLaunchKernelGGL(pe_kernel, dim3(LL), dim3(DD / 2), 0, stream, pe);
  hipLaunchKernelGGL(init_tok_kernel, dim3(1), dim3(256), 0, stream, tok);
  hipLaunchKernelGGL(cvt_h_kernel, dim3((NROWS * DD + 255) / 256), dim3(256), 0, stream,
                     z, zh, NROWS * DD);
  for (int l = 0; l < NLL; ++l) {
    for (int m = 0; m < 4; ++m) {
      hipLaunchKernelGGL(convt_kernel, dim3((DD * DD + 255) / 256), dim3(256), 0, stream,
                         sa_W[m] + l * DD * DD, sa_Wt[m] + l * DD * DD, DD, DD);
      hipLaunchKernelGGL(convt_kernel, dim3((DD * DD + 255) / 256), dim3(256), 0, stream,
                         ca_W[m] + l * DD * DD, ca_Wt[m] + l * DD * DD, DD, DD);
    }
    hipLaunchKernelGGL(convt_kernel, dim3((DD * FFF + 255) / 256), dim3(256), 0, stream,
                       W1 + l * DD * FFF, W1t + l * DD * FFF, DD, FFF);
    hipLaunchKernelGGL(convt_kernel, dim3((FFF * DD + 255) / 256), dim3(256), 0, stream,
                       W2 + l * FFF * DD, W2t + l * FFF * DD, FFF, DD);
  }
  // cross-attention K/V of the fixed memory, per layer (invariant over steps)
  const int gT = (NROWS / 16) * (DD / 16);        // 192 tiles
  const int gB = (gT + 7) / 8;                    // 24 blocks
  for (int l = 0; l < NLL; ++l) {
    hipLaunchKernelGGL(gemm_kernel, dim3(gB), dim3(256), 0, stream,
                       zh, ca_Wt[1] + l * DD * DD, ca_b[1] + l * DD,
                       kmem + l * NROWS * DD, (_Float16*)nullptr, NROWS, DD, DD, 0);
    hipLaunchKernelGGL(gemm_kernel, dim3(gB), dim3(256), 0, stream,
                       zh, ca_Wt[2] + l * DD * DD, ca_b[2] + l * DD,
                       vmem + l * NROWS * DD, (_Float16*)nullptr, NROWS, DD, DD, 0);
  }

  // --- greedy decode loop ---
  for (int t = 0; t < NSTEPS; ++t) {
    hipLaunchKernelGGL(embed_kernel, dim3(NROWS), dim3(DD), 0, stream, tok, emb, pe, x, xh);
    for (int l = 0; l < NLL; ++l) {
      // self-attention: fused Q/K/V, attention, fused O-proj + residual + LN
      hipLaunchKernelGGL(qkv_kernel, dim3((3 * gT + 7) / 8), dim3(256), 0, stream,
                         xh, sa_Wt[0] + l * DD * DD, sa_Wt[1] + l * DD * DD,
                         sa_Wt[2] + l * DD * DD,
                         sa_b[0] + l * DD, sa_b[1] + l * DD, sa_b[2] + l * DD,
                         qb, kb, vb);
      hipLaunchKernelGGL(attn_kernel, dim3(BB * HH), dim3(32), 0, stream, qb, kb, vb, abh, 1);
      hipLaunchKernelGGL(gemm_res_ln_kernel, dim3(NROWS / 16), dim3(512), 0, stream,
                         abh, sa_Wt[3] + l * DD * DD, sa_b[3] + l * DD, x,
                         ln_g[0] + l * DD, ln_b[0] + l * DD, x, xh, DD);
      // cross-attention (K/V precomputed)
      hipLaunchKernelGGL(gemm_kernel, dim3(gB), dim3(256), 0, stream,
                         xh, ca_Wt[0] + l * DD * DD, ca_b[0] + l * DD,
                         qb, (_Float16*)nullptr, NROWS, DD, DD, 0);
      hipLaunchKernelGGL(attn_kernel, dim3(BB * HH), dim3(32), 0, stream,
                         qb, kmem + l * NROWS * DD, vmem + l * NROWS * DD, abh, 0);
      hipLaunchKernelGGL(gemm_res_ln_kernel, dim3(NROWS / 16), dim3(512), 0, stream,
                         abh, ca_Wt[3] + l * DD * DD, ca_b[3] + l * DD, x,
                         ln_g[1] + l * DD, ln_b[1] + l * DD, x, xh, DD);
      // feed-forward: FF1 (ReLU, f16 out) then fused FF2 + residual + LN
      const int fT = (NROWS / 16) * (FFF / 16);   // 768 tiles
      hipLaunchKernelGGL(gemm_kernel, dim3((fT + 7) / 8), dim3(256), 0, stream,
                         xh, W1t + l * DD * FFF, b1 + l * FFF,
                         (float*)nullptr, h1h, NROWS, FFF, DD, 1);
      hipLaunchKernelGGL(gemm_res_ln_kernel, dim3(NROWS / 16), dim3(512), 0, stream,
                         h1h, W2t + l * FFF * DD, b2 + l * DD, x,
                         ln_g[2] + l * DD, ln_b[2] + l * DD, x, xh, FFF);
    }
    hipLaunchKernelGGL(logits_kernel, dim3(BB), dim3(VV), 0, stream,
                       x, Wout, bout, tok, out_logits, out_tok, t);
  }
}
